// int8Net_41051297415510
// MI455X (gfx1250) — matrix operand order repaired and checked by reading, not compiled, not run on hardware
//
#include <hip/hip_runtime.h>
#include <hip/hip_bf16.h>

typedef __attribute__((ext_vector_type(16))) _Float16 v16h;
typedef __attribute__((ext_vector_type(8)))  float    v8f;
typedef __attribute__((ext_vector_type(4)))  unsigned int tdm_g0_t;
typedef __attribute__((ext_vector_type(8)))  int          tdm_g1_t;
typedef __attribute__((ext_vector_type(4)))  int          tdm_g2_t;

#if defined(__gfx1250__) && __has_builtin(__builtin_amdgcn_tensor_load_to_lds)
#define HAVE_TDM 1
#else
#define HAVE_TDM 0
#endif

// ---------------------------------------------------------------------------
// TDM: async 2D f32 tile load Global -> LDS (D# per CDNA5 ISA ch.8).
// Tile = tile_rows x tile_cols elements at gaddr; row stride = colsK elems.
// OOB rows/cols (beyond rowsN/colsK) read as zero by the tensor engine.
// Toolchain note: this clang exposes the 6-arg builtin
//   (u32x4 g0, i32x8 g1, i32x4 g2, i32x4 g3, i32x8 g4, i32 cpol)
// ---------------------------------------------------------------------------
#if HAVE_TDM
__device__ __forceinline__
void tdm_load_2d_f32(unsigned lds_off, const float* gaddr,
                     unsigned rowsN, unsigned colsK,
                     unsigned tile_rows, unsigned tile_cols)
{
    const unsigned long long ga = (unsigned long long)(size_t)gaddr;
    tdm_g0_t g0;
    g0.x = 1u;                                            // count=1 (valid user D#)
    g0.y = lds_off;                                       // lds_addr (bytes)
    g0.z = (unsigned)(ga & 0xffffffffull);                // global_addr[31:0]
    g0.w = (unsigned)((ga >> 32) & 0x01ffffffull)         // global_addr[56:32]
         | (2u << 30);                                    // type=2 ("image")
    tdm_g1_t g1;
    g1[0] = (int)(2u << 16);                              // wg_mask=0, data_size=2 (4B)
    g1[1] = (int)((colsK & 0xffffu) << 16);               // tensor_dim0[15:0]
    g1[2] = (int)((colsK >> 16) | ((rowsN & 0xffffu) << 16)); // dim0 hi | dim1 lo
    g1[3] = (int)((rowsN >> 16) | (tile_cols << 16));     // dim1 hi | tile_dim0
    g1[4] = (int)(tile_rows & 0xffffu);                   // tile_dim1 | tile_dim2=0
    g1[5] = (int)colsK;                                   // tensor_dim0_stride[31:0]
    g1[6] = 0;                                            // stride hi | dim1_stride lo
    g1[7] = 0;
    tdm_g2_t gz4 = {0, 0, 0, 0};                          // 2D: groups 2/3 zero
    tdm_g1_t gz8 = {0, 0, 0, 0, 0, 0, 0, 0};              // trailing group unused
    __builtin_amdgcn_tensor_load_to_lds(g0, g1, gz4, gz4, gz8, 0);
}
#endif

// ---------------------------------------------------------------------------
// conv3x3 + quant epilogue (implicit GEMM, V_WMMA_F32_16X16X32_F16)
// Block: 256 threads = 8 wave32. Tile: 32 (Cout) x 64 (flat N*H*W positions).
// Epilogue: out = relu(acc/(scale[c]*blob) + bias[c])   (== qstage + relu)
// ---------------------------------------------------------------------------
__global__ __launch_bounds__(256)
void conv3x3_qstage_wmma(const float* __restrict__ X,
                         const float* __restrict__ Wt,
                         const float* __restrict__ Bias,
                         const float* __restrict__ Scale,
                         const float* __restrict__ BlobV, int blobIdx,
                         float* __restrict__ Y,
                         int Nb, int Cin, int Cout, int H, int W)
{
    const int K9 = Cin * 9;
    const int HW = H * W;
    const int P  = Nb * HW;

    __shared__ _Float16 As[32][34];   // [m][k]   weights tile
    __shared__ _Float16 Bs[64][34];   // [n][k]   im2col tile (n-major => K contiguous)

    const int t    = threadIdx.x;
    const int wv   = t >> 5;
    const int lane = t & 31;
    const int h    = lane >> 4;       // lane half (wave32)
    const int lm   = lane & 15;

    const int m_base = blockIdx.y * 32;
    const int n_base = blockIdx.x * 64;

    const float blob = BlobV[blobIdx];

    const int am = t >> 3;            // 0..31   (A row)
    const int ak = (t & 7) * 4;       // 4 k's each
    const int bn = t >> 2;            // 0..63   (B row = position)
    const int bk = (t & 3) * 8;       // 8 k's each

    const int bp = n_base + bn;
    int bb = 0, by = 0, bx = 0;
    const bool bvalid = (bp < P);
    if (bvalid) { bb = bp / HW; int rem = bp - bb * HW; by = rem / W; bx = rem - by * W; }

    const int m_off = (wv & 1) * 16;
    const int n_off = (wv >> 1) * 16;

    v8f acc = {};

    const int Ksteps = (K9 + 31) >> 5;
    for (int ks = 0; ks < Ksteps; ++ks) {
        const int k0 = ks << 5;

        // ---- stage A: weights [Cout, Cin*9], f32 -> f16 ----
        {
            const int gm = m_base + am;
            #pragma unroll
            for (int j = 0; j < 4; ++j) {
                const int kk = k0 + ak + j;
                float v = (kk < K9) ? Wt[gm * K9 + kk] : 0.0f;
                As[am][ak + j] = (_Float16)v;
            }
        }
        // ---- stage B: im2col of round(x*blob), zero-padded borders ----
        {
            #pragma unroll
            for (int j = 0; j < 8; ++j) {
                const int kk = k0 + bk + j;
                float v = 0.0f;
                if (bvalid && kk < K9) {
                    const int ci = kk / 9;
                    const int r  = kk - ci * 9;
                    const int ry = r / 3;
                    const int yy = by + ry - 1;
                    const int xx = bx + (r - ry * 3) - 1;
                    if (yy >= 0 && yy < H && xx >= 0 && xx < W)
                        v = __builtin_rintf(X[((bb * Cin + ci) * H + yy) * W + xx] * blob);
                }
                Bs[bn][bk + j] = (_Float16)v;
            }
        }
        __syncthreads();

        // A 16x32 (MxK): m = lane&15; K(e) = (e<8 ? e : e+8) + 8*(lane>>4)
        // B 32x16 (KxN): n = lane&15; K(e) = (e<8 ? e : e+8) + 8*(lane>>4)
        v16h a, b;
        #pragma unroll
        for (int j = 0; j < 8; ++j) {
            const int ka = ((j < 4) ? (2 * j) : (2 * j + 8)) + 8 * h;
            a[2 * j]     = As[m_off + lm][ka];
            a[2 * j + 1] = As[m_off + lm][ka + 1];
            const int kb = ((j < 4) ? (2 * j) : (2 * j + 8)) + 8 * h;
            b[2 * j]     = Bs[n_off + lm][kb];
            b[2 * j + 1] = Bs[n_off + lm][kb + 1];
        }
        acc = __builtin_amdgcn_wmma_f32_16x16x32_f16(false, a, false, b,
                                                     (short)0, acc, false, false);
        __syncthreads();
    }

    // D layout f32 16x16: M = r + 8*h, N = lane&15
    #pragma unroll
    for (int r = 0; r < 8; ++r) {
        const int cout = m_base + m_off + r + 8 * h;
        const int p    = n_base + n_off + lm;
        if (p < P) {
            const float inv = 1.0f / (Scale[cout] * blob);
            float v = acc[r] * inv + Bias[cout];
            v = v > 0.0f ? v : 0.0f;
            const int pb  = p / HW;
            const int rem = p - pb * HW;
            const int py  = rem / W;
            const int px  = rem - py * W;
            Y[((pb * Cout + cout) * H + py) * W + px] = v;
        }
    }
}

// ---------------------------------------------------------------------------
// FC layer: C[M,N] = act(A[M,K] @ W[N,K]^T + bias), WMMA f16->f32.
// Block: 256 threads = 8 waves. Tile: 16 (M=batch rows) x 128 (N).
// Weight tile (128 x 32 f32) streamed into LDS by the Tensor Data Mover
// (TENSORcnt path); OOB rows beyond N auto-zero, so fc3 (N=10) needs no guards.
// ---------------------------------------------------------------------------
__global__ __launch_bounds__(256)
void fc_wmma(const float* __restrict__ A, const float* __restrict__ Wt,
             const float* __restrict__ Bias, float* __restrict__ C,
             int M, int N, int K, int relu)
{
    __shared__ _Float16 As[16][34];
    __shared__ float    Bs32[128][32];    // raw f32 weight tile (TDM target)

    const int t    = threadIdx.x;
    const int wv   = t >> 5;
    const int lane = t & 31;
    const int h    = lane >> 4;
    const int lm   = lane & 15;

    const int m_base = blockIdx.y * 16;
    const int n_base = blockIdx.x * 128;

#if HAVE_TDM
    const unsigned bs_lds = (unsigned)(size_t)&Bs32[0][0];
#endif

    v8f acc = {};

    for (int k0 = 0; k0 < K; k0 += 32) {
        // ---- B tile: 128 rows x 32 cols of f32 weights ----
#if HAVE_TDM
        if (wv == 0) {
            // async DMA: rows [n_base, n_base+128) x cols [k0, k0+32)
            tdm_load_2d_f32(bs_lds, Wt + (size_t)n_base * K + k0,
                            (unsigned)(N - n_base < 0 ? 0 : N - n_base),
                            (unsigned)(K - k0), 128u, 32u);
        }
#else
        {
            const int bnr = t >> 1;
            const int bkk = (t & 1) * 16;
            const int gn  = n_base + bnr;
            const float* wp = Wt + (size_t)gn * K + k0 + bkk;
            #pragma unroll
            for (int j = 0; j < 16; ++j)
                Bs32[bnr][bkk + j] = (gn < N) ? wp[j] : 0.0f;
        }
#endif
        // ---- A tile: 16x32 -> 2 elems/thread, f32 -> f16 ----
        {
            const int am = t >> 4;
            const int ak = (t & 15) * 2;
            const int gm = m_base + am;
            #pragma unroll
            for (int j = 0; j < 2; ++j) {
                float v = (gm < M) ? A[gm * K + k0 + ak + j] : 0.0f;
                As[am][ak + j] = (_Float16)v;
            }
        }
#if HAVE_TDM
        if (wv == 0)
            __builtin_amdgcn_s_wait_tensorcnt(0);
#endif
        __syncthreads();

        v16h a, b;
        const int n_off = wv * 16;
        #pragma unroll
        for (int j = 0; j < 8; ++j) {
            const int ka = ((j < 4) ? (2 * j) : (2 * j + 8)) + 8 * h;
            a[2 * j]     = As[lm][ka];
            a[2 * j + 1] = As[lm][ka + 1];
            const int kb = ((j < 4) ? (2 * j) : (2 * j + 8)) + 8 * h;
            b[2 * j]     = (_Float16)Bs32[n_off + lm][kb];
            b[2 * j + 1] = (_Float16)Bs32[n_off + lm][kb + 1];
        }
        acc = __builtin_amdgcn_wmma_f32_16x16x32_f16(false, a, false, b,
                                                     (short)0, acc, false, false);
        __syncthreads();
    }

    const int col = n_base + wv * 16 + lm;
    if (col < N) {
        const float bias = Bias[col];
        #pragma unroll
        for (int r = 0; r < 8; ++r) {
            const int row = m_base + r + 8 * h;
            if (row < M) {
                float v = acc[r] + bias;
                if (relu) v = v > 0.0f ? v : 0.0f;
                C[(size_t)row * N + col] = v;
            }
        }
    }
}

// ---------------------------------------------------------------------------
// maxpool 2x2 stride 2
// ---------------------------------------------------------------------------
__global__ void maxpool2_kernel(const float* __restrict__ X, float* __restrict__ Y,
                                int NC, int H, int W)
{
    const int Ho = H >> 1, Wo = W >> 1;
    const int total = NC * Ho * Wo;
    int i = blockIdx.x * blockDim.x + threadIdx.x;
    if (i >= total) return;
    int nc = i / (Ho * Wo); int rem = i - nc * (Ho * Wo);
    int y = rem / Wo;       int x = rem - y * Wo;
    const float* p = X + ((size_t)nc * H + 2 * y) * W + 2 * x;
    Y[i] = fmaxf(fmaxf(p[0], p[1]), fmaxf(p[W], p[W + 1]));
}

// ---------------------------------------------------------------------------
// avgpool 2x2 stride 1 (VALID): out (H-1)x(W-1), scaled by 0.25
// ---------------------------------------------------------------------------
__global__ void avgpool2s1_kernel(const float* __restrict__ X, float* __restrict__ Y,
                                  int NC, int H, int W)
{
    const int Ho = H - 1, Wo = W - 1;
    const int total = NC * Ho * Wo;
    int i = blockIdx.x * blockDim.x + threadIdx.x;
    if (i >= total) return;
    int nc = i / (Ho * Wo); int rem = i - nc * (Ho * Wo);
    int y = rem / Wo;       int x = rem - y * Wo;
    const float* p = X + ((size_t)nc * H + y) * W + x;
    Y[i] = 0.25f * (p[0] + p[1] + p[W] + p[W + 1]);
}

// ---------------------------------------------------------------------------
extern "C" void kernel_launch(void* const* d_in, const int* in_sizes, int n_in,
                              void* d_out, int out_size, void* d_ws, size_t ws_size,
                              hipStream_t stream)
{
    const float* x    = (const float*)d_in[0];
    const float* blob = (const float*)d_in[1];
    const float *w[7], *b[7], *s[7];
    for (int i = 0; i < 7; ++i) {
        w[i] = (const float*)d_in[2 + 3 * i];
        b[i] = (const float*)d_in[3 + 3 * i];
        s[i] = (const float*)d_in[4 + 3 * i];
    }
    const float* wfc1 = (const float*)d_in[23]; const float* bfc1 = (const float*)d_in[24];
    const float* wfc2 = (const float*)d_in[25]; const float* bfc2 = (const float*)d_in[26];
    const float* wfc3 = (const float*)d_in[27]; const float* bfc3 = (const float*)d_in[28];
    float* out = (float*)d_out;

    // ping-pong scratch: two halves of d_ws (need >= ~67 MB each)
    char*  ws   = (char*)d_ws;
    size_t half = (ws_size / 2) & ~(size_t)255;
    float* bufA = (float*)ws;
    float* bufB = (float*)(ws + half);

    const int NB = 64;

    auto conv = [&](const float* in, int stage, float* outp,
                    int Cin, int Cout, int H, int W) {
        const int P = NB * H * W;
        dim3 grid((P + 63) / 64, Cout / 32);
        conv3x3_qstage_wmma<<<grid, 256, 0, stream>>>(
            in, w[stage], b[stage], s[stage], blob, stage, outp, NB, Cin, Cout, H, W);
    };
    auto pool = [&](const float* in, float* outp, int C, int H, int W) {
        int total = NB * C * (H / 2) * (W / 2);
        maxpool2_kernel<<<(total + 255) / 256, 256, 0, stream>>>(in, outp, NB * C, H, W);
    };
    auto fc = [&](const float* a, const float* wt, const float* bias, float* c,
                  int M, int N, int K, int relu) {
        dim3 grid((N + 127) / 128, (M + 15) / 16);
        fc_wmma<<<grid, 256, 0, stream>>>(a, wt, bias, c, M, N, K, relu);
    };

    // stage 1: 3 -> 64 @ 64x64
    conv(x, 0, bufA, 3, 64, 64, 64);
    // stage 2: 64 -> 64 @ 64x64, then maxpool -> 32x32
    conv(bufA, 1, bufB, 64, 64, 64, 64);
    pool(bufB, bufA, 64, 64, 64);
    // stage 3: 64 -> 128 @ 32x32
    conv(bufA, 2, bufB, 64, 128, 32, 32);
    // stage 4: 128 -> 128 @ 32x32, maxpool -> 16x16
    conv(bufB, 3, bufA, 128, 128, 32, 32);
    pool(bufA, bufB, 128, 32, 32);
    // stage 5: 128 -> 256 @ 16x16
    conv(bufB, 4, bufA, 128, 256, 16, 16);
    // stage 6: 256 -> 256 @ 16x16
    conv(bufA, 5, bufB, 256, 256, 16, 16);
    // stage 7: 256 -> 256 @ 16x16, maxpool -> 8x8
    conv(bufB, 6, bufA, 256, 256, 16, 16);
    pool(bufA, bufB, 256, 16, 16);
    // avgpool 2x2 s1: 8x8 -> 7x7 (flatten order == NCHW reshape)
    {
        int total = NB * 256 * 7 * 7;
        avgpool2s1_kernel<<<(total + 255) / 256, 256, 0, stream>>>(bufB, bufA, NB * 256, 8, 8);
    }
    // FC head
    fc(bufA, wfc1, bfc1, bufB, 64, 2048, 12544, 1);
    fc(bufB, wfc2, bfc2, bufA, 64, 256,  2048,  1);
    fc(bufA, wfc3, bfc3, out,  64, 10,   256,   0);
}
